// GCNModel_22789096472972
// MI455X (gfx1250) — matrix operand order repaired and checked by reading, not compiled, not run on hardware
//
#include <hip/hip_runtime.h>
#include <hip/hip_bf16.h>

typedef __attribute__((ext_vector_type(16))) _Float16 v16h;
typedef __attribute__((ext_vector_type(8)))  float    v8f;

#define F_IN 128
#define H1   256

// ---------------- small prep kernels ----------------

__global__ void k_init(float* __restrict__ deg, float* __restrict__ z, int n) {
  int i = blockIdx.x * blockDim.x + threadIdx.x;
  if (i < n) { deg[i] = 1.0f; z[i] = 0.0f; }   // deg starts at 1 (self loop)
}

__global__ void k_degree(const int* __restrict__ ei, int E, float* __restrict__ deg) {
  int e = blockIdx.x * blockDim.x + threadIdx.x;
  if (e < E) atomicAdd(&deg[ei[E + e]], 1.0f);  // dst = ei[1][e]
}

__global__ void k_dinv(const float* __restrict__ deg, float* __restrict__ dinv, int n) {
  int i = blockIdx.x * blockDim.x + threadIdx.x;
  if (i < n) dinv[i] = __frsqrt_rn(deg[i]);
}

// wf = W2 @ Wl  (256-vector);  cb = b2 . Wl + bl  (scalar)
__global__ void k_prep(const float* __restrict__ W2, const float* __restrict__ b2,
                       const float* __restrict__ Wl, const float* __restrict__ bl,
                       float* __restrict__ wf, float* __restrict__ cb) {
  int t = threadIdx.x;                 // 256 threads
  float s = 0.f;
  for (int k = 0; k < F_IN; ++k) s += W2[t * F_IN + k] * Wl[k];
  wf[t] = s;
  if (t == 0) {
    float c = 0.f;
    for (int k = 0; k < F_IN; ++k) c += b2[k] * Wl[k];
    cb[0] = c + bl[0];
  }
}

// Pack W1 (f32 [128][256]) into per-lane WMMA B fragments (f16), so the GEMM
// reads each 16-half fragment as one aligned 32-byte load.
// Bpack[(((kk*16 + ntile)*32 + lane)*16) + e] = W1[K][n]
//   K = kk*32 + 8*(lane>>4) + (e&7) + 16*(e>>3) ;  n = ntile*16 + (lane&15)
__global__ void k_bpack(const float* __restrict__ W1, _Float16* __restrict__ Bpack) {
  int idx = blockIdx.x * blockDim.x + threadIdx.x;   // 4*16*32*16 = 32768
  if (idx < F_IN * H1) {
    int e     = idx & 15;
    int lane  = (idx >> 4) & 31;
    int ntile = (idx >> 9) & 15;
    int kk    = idx >> 13;
    int K     = kk * 32 + ((lane >> 4) << 3) + (e & 7) + ((e >> 3) << 4);
    int n     = ntile * 16 + (lane & 15);
    Bpack[idx] = (_Float16)W1[K * H1 + n];
  }
}

// agg[i] = dinv[i]^2 * x[i]   (self-loop term, also zero-initializes agg)
__global__ void k_selfinit(const float* __restrict__ x, const float* __restrict__ dinv,
                           float* __restrict__ agg, int n) {
  int idx = blockIdx.x * blockDim.x + threadIdx.x;     // n * 32 float4 slots
  if (idx < n * 32) {
    int node = idx >> 5;
    float dv = dinv[node]; dv *= dv;
    float4 v = ((const float4*)x)[idx];
    float4 o; o.x = dv * v.x; o.y = dv * v.y; o.z = dv * v.z; o.w = dv * v.w;
    ((float4*)agg)[idx] = o;
  }
}

// 128-wide edge aggregation of x: 32 lanes per edge, one float4 per lane.
__global__ void k_scatter1(const int* __restrict__ ei, int E,
                           const float* __restrict__ dinv,
                           const float* __restrict__ x, float* __restrict__ agg) {
  long long gid = (long long)blockIdx.x * blockDim.x + threadIdx.x;
  int e = (int)(gid >> 5);
  int lane = (int)(gid & 31);
  if (e < E) {
    int s = ei[e];
    int d = ei[E + e];
    float nrm = dinv[s] * dinv[d];
    float4 v = ((const float4*)x)[(size_t)s * 32 + lane];
    float* p = agg + (size_t)d * F_IN + lane * 4;
    atomicAdd(p + 0, nrm * v.x);
    atomicAdd(p + 1, nrm * v.y);
    atomicAdd(p + 2, nrm * v.z);
    atomicAdd(p + 3, nrm * v.w);
  }
}

// ---------------- WMMA GEMM + fused head ----------------
// h1 = relu(agg @ W1 + b1) on a 32-row x 256-col stripe per block, then
// z[m] += h1[m,:].wf fused in the epilogue.
// Block: 128 threads = 4 waves. Wave w owns columns [w*64, w*64+64) (4 n-tiles)
// and both 16-row m-tiles, so every B fragment feeds two WMMAs.
__global__ __launch_bounds__(128) void k_gemm_fused(
    const float*    __restrict__ agg,
    const _Float16* __restrict__ Bpack,
    const float*    __restrict__ b1,
    const float*    __restrict__ wf,
    float*          __restrict__ z) {
  const int wave  = threadIdx.x >> 5;
  const int lane  = threadIdx.x & 31;
  const int khalf = lane >> 4;       // 0: K-low chunk lanes, 1: K-high
  const int sub   = lane & 15;       // row (A) / col (B,C)
  const int m0    = blockIdx.x * 32; // two 16-row tiles

  v8f acc[2][4] = {};                // [m-tile][n-tile] 16x16 f32 accumulators
  const float* Arow0 = agg + (size_t)(m0 + sub) * F_IN;
  const float* Arow1 = Arow0 + (size_t)16 * F_IN;

  #pragma unroll
  for (int kk = 0; kk < 4; ++kk) {   // K = 128 in steps of 32
    // A fragments (16-bit A 16x32 layout): contiguous 8-f32 chunks -> b128 loads
    const int kbA = kk * 32 + khalf * 8;
    v16h a0, a1;
    #pragma unroll
    for (int e = 0; e < 8; ++e) {
      a0[e]     = (_Float16)Arow0[kbA + e];
      a0[8 + e] = (_Float16)Arow0[kbA + 16 + e];
      a1[e]     = (_Float16)Arow1[kbA + e];
      a1[8 + e] = (_Float16)Arow1[kbA + 16 + e];
    }
    #pragma unroll
    for (int t = 0; t < 4; ++t) {
      const int ntile = wave * 4 + t;
      // one aligned 32B fragment load (2x global_load_b128)
      v16h b = *(const v16h*)(Bpack + (((size_t)kk * 16 + ntile) * 32 + lane) * 16);
      acc[0][t] = __builtin_amdgcn_wmma_f32_16x16x32_f16(
          false, a0, false, b, (short)0, acc[0][t], false, false);
      acc[1][t] = __builtin_amdgcn_wmma_f32_16x16x32_f16(
          false, a1, false, b, (short)0, acc[1][t], false, false);
    }
  }

  // Epilogue: bias + relu, then fused z[m] += sum_n h1[m][n]*wf[n]
  // C/D layout: VGPR r holds (m = mbase + r + 8*khalf, n = wave*64 + t*16 + sub)
  #pragma unroll
  for (int mi = 0; mi < 2; ++mi) {
    const int mbase = m0 + mi * 16;
    #pragma unroll
    for (int t = 0; t < 4; ++t) {
      const int ncol = wave * 64 + t * 16 + sub;
      const float bias = b1[ncol];
      const float wn   = wf[ncol];
      #pragma unroll
      for (int r = 0; r < 8; ++r) {
        float h = acc[mi][t][r] + bias;
        h = h > 0.f ? h : 0.f;          // relu
        float s = h * wn;
        s += __shfl_xor(s, 1);          // butterfly over the 16 lanes sharing m
        s += __shfl_xor(s, 2);
        s += __shfl_xor(s, 4);
        s += __shfl_xor(s, 8);
        if (sub == 0) atomicAdd(&z[mbase + r + khalf * 8], s);
      }
    }
  }
}

// ---------------- collapsed layer-2 aggregation (1-wide) ----------------

__global__ void k_out_self(const float* __restrict__ dinv, const float* __restrict__ zv,
                           const float* __restrict__ cb, float* __restrict__ out, int n) {
  int i = blockIdx.x * blockDim.x + threadIdx.x;
  if (i < n) {
    float dv = dinv[i];
    out[i] = dv * dv * zv[i] + cb[0];
  }
}

__global__ void k_scatter2(const int* __restrict__ ei, int E,
                           const float* __restrict__ dinv,
                           const float* __restrict__ zv, float* __restrict__ out) {
  int e = blockIdx.x * blockDim.x + threadIdx.x;
  if (e < E) {
    int s = ei[e];
    int d = ei[E + e];
    atomicAdd(&out[d], dinv[s] * dinv[d] * zv[s]);
  }
}

// ---------------- launcher ----------------

extern "C" void kernel_launch(void* const* d_in, const int* in_sizes, int n_in,
                              void* d_out, int out_size, void* d_ws, size_t ws_size,
                              hipStream_t stream) {
  (void)n_in; (void)out_size; (void)ws_size;
  const float* x  = (const float*)d_in[0];
  const int*   ei = (const int*)d_in[1];
  const float* W1 = (const float*)d_in[2];
  const float* b1 = (const float*)d_in[3];
  const float* W2 = (const float*)d_in[4];
  const float* b2 = (const float*)d_in[5];
  const float* Wl = (const float*)d_in[6];
  const float* bl = (const float*)d_in[7];
  float* out = (float*)d_out;

  const int N = in_sizes[0] / F_IN;   // 100000
  const int E = in_sizes[1] / 2;      // 1600000

  char* ws = (char*)d_ws;
  float*    deg   = (float*)(ws);
  float*    dinv  = (float*)(ws + (size_t)N * 4);
  float*    zv    = (float*)(ws + (size_t)N * 8);
  float*    wf    = (float*)(ws + (size_t)N * 12);
  float*    cb    = wf + H1;
  _Float16* Bpack = (_Float16*)(ws + (size_t)N * 12 + 2048);
  float*    agg   = (float*)(ws + (size_t)N * 12 + 2048 + (size_t)F_IN * H1 * 2);

  const int TB = 256;
  k_init    <<<(N + TB - 1) / TB, TB, 0, stream>>>(deg, zv, N);
  k_degree  <<<(E + TB - 1) / TB, TB, 0, stream>>>(ei, E, deg);
  k_dinv    <<<(N + TB - 1) / TB, TB, 0, stream>>>(deg, dinv, N);
  k_prep    <<<1, H1, 0, stream>>>(W2, b2, Wl, bl, wf, cb);
  k_bpack   <<<(F_IN * H1 + TB - 1) / TB, TB, 0, stream>>>(W1, Bpack);
  k_selfinit<<<((long long)N * 32 + TB - 1) / TB, TB, 0, stream>>>(x, dinv, agg, N);
  {
    long long tot = (long long)E * 32;
    k_scatter1<<<(unsigned)((tot + TB - 1) / TB), TB, 0, stream>>>(ei, E, dinv, x, agg);
  }
  k_gemm_fused<<<N / 32, 128, 0, stream>>>(agg, Bpack, b1, wf, zv);
  k_out_self<<<(N + TB - 1) / TB, TB, 0, stream>>>(dinv, zv, cb, out, N);
  k_scatter2<<<(E + TB - 1) / TB, TB, 0, stream>>>(ei, E, dinv, zv, out);
}
